// EncoderBlock_17239998726158
// MI455X (gfx1250) — compile-verified, matrix operand order checked by reading, not run on hardware
//
#include <hip/hip_runtime.h>
#include <hip/hip_bf16.h>

// ---------------------------------------------------------------------------
// EncoderBlock for MI455X (gfx1250, wave32, WMMA + Tensor Data Mover).
// bf16 WMMA (v_wmma_f32_16x16x32_bf16) for all matmuls, f32 accumulate.
// Flash-attention (online softmax) so the [B,H,S,S] score tensor never
// touches HBM. GEMM tiles staged by TDM (tensor_load_to_lds, TENSORcnt),
// double-buffered, so data movement stays off the VALU/DS pipes.
// ---------------------------------------------------------------------------

typedef unsigned short u16;
typedef __attribute__((ext_vector_type(16))) __bf16 v16bf;
typedef __attribute__((ext_vector_type(8)))  float  v8f;

#if defined(__has_builtin)
#if __has_builtin(__builtin_amdgcn_tensor_load_to_lds) && __has_builtin(__builtin_amdgcn_s_wait_tensorcnt)
#define HAVE_TDM 1
#endif
#endif
#ifndef HAVE_TDM
#define HAVE_TDM 0
#endif

union Frag16 {            // one WMMA 16-bit A/B operand: 16 bf16 = 8 VGPRs
    v16bf v;
    uint4 q[2];
    u16   s[16];
};

__device__ __forceinline__ u16 f2bf(float f) {
    union { float f; unsigned u; } c; c.f = f;
    unsigned r = c.u + 0x7fffu + ((c.u >> 16) & 1u);   // round-to-nearest-even
    return (u16)(r >> 16);
}

__device__ __forceinline__ float gelu_new(float x) {
    const float c = 0.7978845608028654f;               // sqrt(2/pi)
    float x3 = x * x * x;
    return 0.5f * x * (1.0f + tanhf(c * (x + 0.044715f * x3)));
}

// low 32 bits of a generic pointer to LDS == LDS byte address (ISA aperture rule)
__device__ __forceinline__ unsigned lds_addr32(const void* p) {
    return (unsigned)(unsigned long long)p;
}

#if HAVE_TDM
// ---------------------------------------------------------------------------
// Issue one TDM 2D tile load: global (row-major, row_stride elems of 2B)
// -> LDS at lds_addr. Optional LDS row padding via pad fields (D# group1).
// D# packing per CDNA5 ISA 8.3/8.4 (group0: count/lds/global/type=2;
// group1: data_size=2B, dims, strides, pad config). 2D => groups 2/3 zero.
// This toolchain exposes the 6-arg builtin:
//   (uint32x4 g0, int32x8 g1, int32x4, int32x4, int32x8, i32 cpol)
// ---------------------------------------------------------------------------
__device__ __forceinline__ void tdm_load_2d(unsigned lds_addr, const void* gaddr,
                                            unsigned tile_d0, unsigned tile_d1,
                                            unsigned tensor_d0, unsigned tensor_d1,
                                            unsigned long long row_stride,
                                            unsigned pad_cfg) {
    typedef unsigned int v4u __attribute__((ext_vector_type(4)));
    typedef int v8i __attribute__((ext_vector_type(8)));
    typedef int v4i __attribute__((ext_vector_type(4)));
    unsigned long long ga = (unsigned long long)gaddr;
    v4u g0;
    g0.x = 1u;                                                    // count=1
    g0.y = lds_addr;                                              // lds_addr[31:0]
    g0.z = (unsigned)ga;                                          // global[31:0]
    g0.w = (unsigned)((ga >> 32) & 0x1FFFFFFu) | (2u << 30);      // global[56:32] | type=2
    v8i g1;
    g1[0] = (int)((1u << 16) | pad_cfg);                          // data_size=2B + pad cfg
    g1[1] = (int)((tensor_d0 & 0xFFFFu) << 16);                   // tensor_dim0[15:0]
    g1[2] = (int)((tensor_d0 >> 16) | ((tensor_d1 & 0xFFFFu) << 16));
    g1[3] = (int)((tensor_d1 >> 16) | (tile_d0 << 16));           // | tile_dim0
    g1[4] = (int)(tile_d1 & 0xFFFFu);                             // tile_dim1 (tile_dim2=0)
    g1[5] = (int)(unsigned)(row_stride & 0xFFFFFFFFull);          // dim0_stride[31:0]
    g1[6] = (int)(unsigned)((row_stride >> 32) & 0xFFFFull);      // dim0_stride[47:32]
    g1[7] = 0;
    v4i z4 = {0, 0, 0, 0};
    v8i z8 = {0, 0, 0, 0, 0, 0, 0, 0};
    __builtin_amdgcn_tensor_load_to_lds(g0, g1, z4, z4, z8, 0);
}
#endif

// ---------------------------------------------------------------------------
// Weight prep: fp32 [K][N] -> bf16 [N][K]  (so B-fragments read contiguous k)
// ---------------------------------------------------------------------------
__global__ __launch_bounds__(256)
void transpose_cvt(const float* __restrict__ W, u16* __restrict__ Wt, int K, int N) {
    __shared__ float t[32][33];
    const int tx = threadIdx.x & 31, ty = threadIdx.x >> 5;
    const int n0 = blockIdx.x * 32, k0 = blockIdx.y * 32;
#pragma unroll
    for (int i = 0; i < 4; ++i)
        t[ty + 8 * i][tx] = W[(size_t)(k0 + ty + 8 * i) * N + n0 + tx];
    __syncthreads();
#pragma unroll
    for (int i = 0; i < 4; ++i)
        Wt[(size_t)(n0 + ty + 8 * i) * K + k0 + tx] = f2bf(t[tx][ty + 8 * i]);
}

// ---------------------------------------------------------------------------
// LayerNorm over D=2048, one 256-thread block per row, bf16 output
// ---------------------------------------------------------------------------
__global__ __launch_bounds__(256)
void layernorm_bf16(const float* __restrict__ x, const float* __restrict__ w,
                    const float* __restrict__ b, u16* __restrict__ out, int D) {
    __shared__ float red[16];
    const int row = blockIdx.x, tid = threadIdx.x;
    const float* xr = x + (size_t)row * D;
    float r[8], s = 0.f, s2 = 0.f;
#pragma unroll
    for (int i = 0; i < 8; ++i) {                      // D/256 == 8
        float t = xr[tid + 256 * i];
        r[i] = t; s += t; s2 += t * t;
    }
#pragma unroll
    for (int m = 16; m >= 1; m >>= 1) { s += __shfl_xor(s, m, 32); s2 += __shfl_xor(s2, m, 32); }
    const int wid = tid >> 5, lane = tid & 31;
    if (lane == 0) { red[wid] = s; red[8 + wid] = s2; }
    __syncthreads();
    s = 0.f; s2 = 0.f;
#pragma unroll
    for (int i = 0; i < 8; ++i) { s += red[i]; s2 += red[8 + i]; }
    const float mu  = s / (float)D;
    const float var = s2 / (float)D - mu * mu;
    const float rs  = rsqrtf(var + 1e-5f);
#pragma unroll
    for (int i = 0; i < 8; ++i) {
        int c = tid + 256 * i;
        out[(size_t)row * D + c] = f2bf((r[i] - mu) * rs * w[c] + b[c]);
    }
}

// ---------------------------------------------------------------------------
// Generic bf16 WMMA GEMM:  C[M,N] = epilogue(A[M,K] @ Bt[N,K]^T)
// 128x128 block tile, BK=32 (one WMMA K-step), 8 waves of 64x32.
// Tiles staged by TDM (double-buffered); wave 0 drives the DMA engine and
// throttles with s_wait_tensorcnt while other waves only compute.
// ---------------------------------------------------------------------------
enum { F_BIAS = 1, F_GELU = 2, F_RES = 4, F_OBF16 = 8, F_SCATTER = 16 };

template <int FLAGS>
__global__ __launch_bounds__(256)
void gemm_bf16(const u16* __restrict__ A, const u16* __restrict__ Bt,
               const float* __restrict__ bias, const float* __restrict__ res,
               float* __restrict__ Cf, u16* __restrict__ Cb,
               int M, int N, int K, int Sdim, int Hn) {
    constexpr int BM = 128, BN = 128, BKc = 32, LDA = BKc + 8;   // 80B rows, 16B aligned
    __shared__ __align__(16) u16 As[2][BM * LDA];
    __shared__ __align__(16) u16 Bs[2][BN * LDA];

    const int tid  = threadIdx.x;
    const int lane = tid & 31, wid = tid >> 5;
    const int bm = blockIdx.y * BM, bn = blockIdx.x * BN;
    const int wm = (wid >> 2) * 64, wn = (wid & 3) * 32;
    const int rl = lane & 15, hl = lane >> 4;

    const v8f vz = {0.f, 0.f, 0.f, 0.f, 0.f, 0.f, 0.f, 0.f};
    v8f acc[4][2];
#pragma unroll
    for (int i = 0; i < 4; ++i)
#pragma unroll
        for (int j = 0; j < 2; ++j) acc[i][j] = vz;

    const int nk = K / BKc;

#if HAVE_TDM
    // pad: rows are 16 DWORDs (32 bf16), insert 4 DWORDs (8 elems) -> stride 40
    const unsigned padcfg = (1u << 20) | (3u << 22) | (3u << 25); // en, ivl=16dw, amt=4dw
    const unsigned la0 = lds_addr32(&As[0][0]), la1 = lds_addr32(&As[1][0]);
    const unsigned lb0 = lds_addr32(&Bs[0][0]), lb1 = lds_addr32(&Bs[1][0]);
    if (wid == 0) {
        tdm_load_2d(la0, A  + (size_t)bm * K, BKc, BM, (unsigned)K, (unsigned)M,
                    (unsigned long long)K, padcfg);
        tdm_load_2d(lb0, Bt + (size_t)bn * K, BKc, BN, (unsigned)K, (unsigned)N,
                    (unsigned long long)K, padcfg);
    }
#endif

    for (int kk = 0; kk < nk; ++kk) {
        const int buf = kk & 1;
#if HAVE_TDM
        if (wid == 0) {
            if (kk + 1 < nk) {                      // prefetch next pair, then wait
                const int k1 = (kk + 1) * BKc;      // for the previous pair only
                tdm_load_2d(buf ? la0 : la1, A  + (size_t)bm * K + k1, BKc, BM,
                            (unsigned)K, (unsigned)M, (unsigned long long)K, padcfg);
                tdm_load_2d(buf ? lb0 : lb1, Bt + (size_t)bn * K + k1, BKc, BN,
                            (unsigned)K, (unsigned)N, (unsigned long long)K, padcfg);
                __builtin_amdgcn_s_wait_tensorcnt(2);
            } else {
                __builtin_amdgcn_s_wait_tensorcnt(0);
            }
        }
        __syncthreads();
#else
        const int k0 = kk * BKc;
#pragma unroll
        for (int i = 0; i < 2; ++i) {               // fallback: manual staging
            int e = (tid + i * 256) * 8;
            int row = e >> 5, col = e & 31;
            *(uint4*)&As[buf][row * LDA + col] =
                *(const uint4*)&A [(size_t)(bm + row) * K + k0 + col];
            *(uint4*)&Bs[buf][row * LDA + col] =
                *(const uint4*)&Bt[(size_t)(bn + row) * K + k0 + col];
        }
        __syncthreads();
#endif
        Frag16 bfr[2];
#pragma unroll
        for (int j = 0; j < 2; ++j) {
            bfr[j].q[0] = *(const uint4*)&Bs[buf][(wn + 16 * j + rl) * LDA + 8 * hl];
            bfr[j].q[1] = *(const uint4*)&Bs[buf][(wn + 16 * j + rl) * LDA + 16 + 8 * hl];
        }
#pragma unroll
        for (int i = 0; i < 4; ++i) {
            Frag16 af;
            af.q[0] = *(const uint4*)&As[buf][(wm + 16 * i + rl) * LDA + 8 * hl];
            af.q[1] = *(const uint4*)&As[buf][(wm + 16 * i + rl) * LDA + 16 + 8 * hl];
#pragma unroll
            for (int j = 0; j < 2; ++j)
                acc[i][j] = __builtin_amdgcn_wmma_f32_16x16x32_bf16(
                    false, af.v, false, bfr[j].v, (short)0, acc[i][j], false, false);
        }
        __syncthreads();
    }

#pragma unroll
    for (int i = 0; i < 4; ++i)
#pragma unroll
        for (int j = 0; j < 2; ++j)
#pragma unroll
            for (int vv = 0; vv < 8; ++vv) {
                int row = bm + wm + 16 * i + vv + 8 * hl;      // C layout: vgpr=row
                int col = bn + wn + 16 * j + rl;               //           lane=col
                float xv = acc[i][j][vv];
                if (FLAGS & F_BIAS) xv += bias[col];
                if (FLAGS & F_GELU) xv = gelu_new(xv);
                if (FLAGS & F_RES)  xv += res[(size_t)row * N + col];
                if (FLAGS & F_OBF16) {
                    if (FLAGS & F_SCATTER) {                   // -> [B,H,S,HD], HD=128
                        int b_ = row / Sdim, sr = row % Sdim;
                        int hh = col >> 7, hd = col & 127;
                        Cb[((size_t)(b_ * Hn + hh) * Sdim + sr) * 128 + hd] = f2bf(xv);
                    } else {
                        Cb[(size_t)row * N + col] = f2bf(xv);
                    }
                } else {
                    Cf[(size_t)row * N + col] = xv;
                }
            }
}

// ---------------------------------------------------------------------------
// Flash attention: one block per (b*H, 128 q rows). HD=128 fixed.
// 8 waves, each owning 16 q rows. Online softmax, scores stay on-chip.
// LDS: Ks[128][128] (aliased by P after a barrier) + Vt[128][128] = 64 KB.
// ---------------------------------------------------------------------------
__global__ __launch_bounds__(256)
void flash_attn(const u16* __restrict__ q, const u16* __restrict__ k,
                const u16* __restrict__ v, u16* __restrict__ o,
                int S, int Hn, float scale) {
    constexpr int HD = 128, LDK = 128;
    __shared__ __align__(16) u16 smem[2 * 128 * LDK];
    u16* Ks = smem;
    u16* Vt = smem + 128 * LDK;

    const int tid = threadIdx.x, lane = tid & 31, wid = tid >> 5;
    const int rl = lane & 15, hl = lane >> 4;
    const int bh = blockIdx.y;
    const int q0 = blockIdx.x * 128;
    const size_t base = (size_t)bh * S * HD;

    // Q fragments (A-operand) straight from global, kept in registers
    Frag16 qf[4];
    const size_t qoff = base + (size_t)(q0 + 16 * wid + rl) * HD;
#pragma unroll
    for (int kb = 0; kb < 4; ++kb) {
        qf[kb].q[0] = *(const uint4*)&q[qoff + 32 * kb + 8 * hl];
        qf[kb].q[1] = *(const uint4*)&q[qoff + 32 * kb + 16 + 8 * hl];
    }

    const v8f vz = {0.f, 0.f, 0.f, 0.f, 0.f, 0.f, 0.f, 0.f};
    v8f oacc[8];
    float m_i[8], l_i[8];
#pragma unroll
    for (int i = 0; i < 8; ++i) { oacc[i] = vz; m_i[i] = -1e30f; l_i[i] = 0.f; }

    for (int j0 = 0; j0 < S; j0 += 128) {
        // stage K row-major, V transposed ([hd][kv])
#pragma unroll
        for (int it = 0; it < 8; ++it) {
            int e = (tid + it * 256) * 8;
            int row = e >> 7, col = e & 127;
            *(uint4*)&Ks[row * LDK + col] =
                *(const uint4*)&k[base + (size_t)(j0 + row) * HD + col];
            uint4 vd = *(const uint4*)&v[base + (size_t)(j0 + row) * HD + col];
            const u16* vs = (const u16*)&vd;
#pragma unroll
            for (int c = 0; c < 8; ++c) Vt[(col + c) * LDK + row] = vs[c];
        }
        __syncthreads();

        // S = Q @ K^T  (K tile's natural layout == B-fragment layout)
        v8f sacc[8];
#pragma unroll
        for (int nt = 0; nt < 8; ++nt) sacc[nt] = vz;
#pragma unroll
        for (int nt = 0; nt < 8; ++nt)
#pragma unroll
            for (int kb = 0; kb < 4; ++kb) {
                Frag16 kf;
                kf.q[0] = *(const uint4*)&Ks[(16 * nt + rl) * LDK + 32 * kb + 8 * hl];
                kf.q[1] = *(const uint4*)&Ks[(16 * nt + rl) * LDK + 32 * kb + 16 + 8 * hl];
                sacc[nt] = __builtin_amdgcn_wmma_f32_16x16x32_bf16(
                    false, qf[kb].v, false, kf.v, (short)0, sacc[nt], false, false);
            }

        // online softmax: row r = vv + 8*hl lives in 16 lanes of one half
#pragma unroll
        for (int vv = 0; vv < 8; ++vv) {
            float mx = -1e30f;
#pragma unroll
            for (int nt = 0; nt < 8; ++nt) {
                float t = sacc[nt][vv] * scale;
                sacc[nt][vv] = t;
                mx = fmaxf(mx, t);
            }
#pragma unroll
            for (int m = 8; m >= 1; m >>= 1) mx = fmaxf(mx, __shfl_xor(mx, m, 32));
            float mnew = fmaxf(m_i[vv], mx);
            float corr = __expf(m_i[vv] - mnew);
            m_i[vv] = mnew;
            float sum = 0.f;
#pragma unroll
            for (int nt = 0; nt < 8; ++nt) {
                float p = __expf(sacc[nt][vv] - mnew);
                sacc[nt][vv] = p;
                sum += p;
            }
#pragma unroll
            for (int m = 8; m >= 1; m >>= 1) sum += __shfl_xor(sum, m, 32);
            l_i[vv] = l_i[vv] * corr + sum;
#pragma unroll
            for (int ht = 0; ht < 8; ++ht) oacc[ht][vv] *= corr;
        }

        __syncthreads();                 // all waves done reading Ks -> alias as P
        u16* P = Ks + wid * 16 * LDK;    // per-wave [16 q rows][128 kv], row-major
#pragma unroll
        for (int nt = 0; nt < 8; ++nt)
#pragma unroll
            for (int vv = 0; vv < 8; ++vv)
                P[(vv + 8 * hl) * LDK + 16 * nt + rl] = f2bf(sacc[nt][vv]);

        // O += P @ V   (Vt transposed -> B-fragment layout)
#pragma unroll
        for (int kb = 0; kb < 4; ++kb) {
            Frag16 pf;
            pf.q[0] = *(const uint4*)&P[rl * LDK + 32 * kb + 8 * hl];
            pf.q[1] = *(const uint4*)&P[rl * LDK + 32 * kb + 16 + 8 * hl];
#pragma unroll
            for (int ht = 0; ht < 8; ++ht) {
                Frag16 vf;
                vf.q[0] = *(const uint4*)&Vt[(16 * ht + rl) * LDK + 32 * kb + 8 * hl];
                vf.q[1] = *(const uint4*)&Vt[(16 * ht + rl) * LDK + 32 * kb + 16 + 8 * hl];
                oacc[ht] = __builtin_amdgcn_wmma_f32_16x16x32_bf16(
                    false, pf.v, false, vf.v, (short)0, oacc[ht], false, false);
            }
        }
        __syncthreads();
    }

    // write O back as [b*S+s][h*HD+hd] (row-major [8192, 2048] bf16)
    const int b_ = bh / Hn, hh = bh % Hn;
    const int D = Hn * HD;
#pragma unroll
    for (int vv = 0; vv < 8; ++vv) {
        float inv = 1.0f / l_i[vv];
        int row = q0 + 16 * wid + vv + 8 * hl;
#pragma unroll
        for (int ht = 0; ht < 8; ++ht) {
            int col = hh * HD + 16 * ht + rl;
            o[(size_t)(b_ * S + row) * D + col] = f2bf(oacc[ht][vv] * inv);
        }
    }
}

// ---------------------------------------------------------------------------
// Launch
// ---------------------------------------------------------------------------
extern "C" void kernel_launch(void* const* d_in, const int* in_sizes, int n_in,
                              void* d_out, int out_size, void* d_ws, size_t ws_size,
                              hipStream_t stream) {
    (void)in_sizes; (void)n_in; (void)out_size; (void)ws_size;
    constexpr int Bb = 4, S = 2048, D = 2048, H = 16, DFF = 8192;
    constexpr int M = Bb * S;                       // 8192 tokens
    const float scale = 0.08838834764831845f;       // 1/sqrt(128)

    const float* x    = (const float*)d_in[0];
    const float* wq   = (const float*)d_in[1];
    const float* wk   = (const float*)d_in[2];
    const float* wv   = (const float*)d_in[3];
    const float* wo   = (const float*)d_in[4];
    const float* ln1w = (const float*)d_in[5];
    const float* ln1b = (const float*)d_in[6];
    const float* ln2w = (const float*)d_in[7];
    const float* ln2b = (const float*)d_in[8];
    const float* fc1w = (const float*)d_in[9];
    const float* fc1b = (const float*)d_in[10];
    const float* fc2w = (const float*)d_in[11];
    const float* fc2b = (const float*)d_in[12];

    // workspace carve-up (~480 MB total)
    char* p = (char*)d_ws;
    auto take = [&](size_t bytes) { char* r = p; p += (bytes + 255) & ~(size_t)255; return r; };
    u16*  wqT  = (u16*)take((size_t)D * D * 2);
    u16*  wkT  = (u16*)take((size_t)D * D * 2);
    u16*  wvT  = (u16*)take((size_t)D * D * 2);
    u16*  woT  = (u16*)take((size_t)D * D * 2);
    u16*  fc1T = (u16*)take((size_t)D * DFF * 2);
    u16*  fc2T = (u16*)take((size_t)DFF * D * 2);
    u16*  h1   = (u16*)take((size_t)M * D * 2);
    u16*  qb   = (u16*)take((size_t)M * D * 2);
    u16*  kb   = (u16*)take((size_t)M * D * 2);
    u16*  vb   = (u16*)take((size_t)M * D * 2);
    u16*  ao   = (u16*)take((size_t)M * D * 2);
    float* x1  = (float*)take((size_t)M * D * 4);
    u16*  h2   = (u16*)take((size_t)M * D * 2);
    u16*  mid  = (u16*)take((size_t)M * DFF * 2);

    // weights -> bf16 transposed [N][K]
    transpose_cvt<<<dim3(D / 32, D / 32),   256, 0, stream>>>(wq,   wqT,  D,   D);
    transpose_cvt<<<dim3(D / 32, D / 32),   256, 0, stream>>>(wk,   wkT,  D,   D);
    transpose_cvt<<<dim3(D / 32, D / 32),   256, 0, stream>>>(wv,   wvT,  D,   D);
    transpose_cvt<<<dim3(D / 32, D / 32),   256, 0, stream>>>(wo,   woT,  D,   D);
    transpose_cvt<<<dim3(DFF / 32, D / 32), 256, 0, stream>>>(fc1w, fc1T, D,   DFF);
    transpose_cvt<<<dim3(D / 32, DFF / 32), 256, 0, stream>>>(fc2w, fc2T, DFF, D);

    // pre-norm MHA
    layernorm_bf16<<<M, 256, 0, stream>>>(x, ln1w, ln1b, h1, D);
    gemm_bf16<F_OBF16 | F_SCATTER><<<dim3(D / 128, M / 128), 256, 0, stream>>>(
        h1, wqT, nullptr, nullptr, nullptr, qb, M, D, D, S, H);
    gemm_bf16<F_OBF16 | F_SCATTER><<<dim3(D / 128, M / 128), 256, 0, stream>>>(
        h1, wkT, nullptr, nullptr, nullptr, kb, M, D, D, S, H);
    gemm_bf16<F_OBF16 | F_SCATTER><<<dim3(D / 128, M / 128), 256, 0, stream>>>(
        h1, wvT, nullptr, nullptr, nullptr, vb, M, D, D, S, H);
    flash_attn<<<dim3(S / 128, Bb * H), 256, 0, stream>>>(qb, kb, vb, ao, S, H, scale);
    gemm_bf16<F_RES><<<dim3(D / 128, M / 128), 256, 0, stream>>>(
        ao, woT, nullptr, x, x1, nullptr, M, D, D, S, H);

    // pre-norm MLP (residual fused into fc2 epilogue, writes d_out fp32)
    layernorm_bf16<<<M, 256, 0, stream>>>(x1, ln2w, ln2b, h2, D);
    gemm_bf16<F_BIAS | F_GELU | F_OBF16><<<dim3(DFF / 128, M / 128), 256, 0, stream>>>(
        h2, fc1T, fc1b, nullptr, nullptr, mid, M, DFF, D, S, H);
    gemm_bf16<F_BIAS | F_RES><<<dim3(D / 128, M / 128), 256, 0, stream>>>(
        mid, fc2T, fc2b, x1, (float*)d_out, nullptr, M, D, DFF, S, H);
}
